// NGCF_pyg_36730560315653
// MI455X (gfx1250) — matrix operand order repaired and checked by reading, not compiled, not run on hardware
//
#include <hip/hip_runtime.h>
#include <hip/hip_bf16.h>

typedef __attribute__((ext_vector_type(16))) _Float16 v16h;
typedef __attribute__((ext_vector_type(8)))  float    v8f;

#define N_NODES   100000
#define EMB       128
#define NUM_EDGES 600000
#define OUT_STRIDE 512
#define LRELU_SLOPE 0.01f
#define L2_EPS 1e-12f

// ---------------- degree kernels ----------------
__global__ void deg_init_kernel(float* __restrict__ deg) {
    int i = blockIdx.x * blockDim.x + threadIdx.x;
    if (i < N_NODES) deg[i] = 1.0f;   // self loop
}

__global__ void deg_edges_kernel(const int* __restrict__ dst, float* __restrict__ deg) {
    int e = blockIdx.x * blockDim.x + threadIdx.x;
    if (e < NUM_EDGES) atomicAdd(&deg[dst[e]], 1.0f);
}

__global__ void deg_finish_kernel(float* __restrict__ dinv) {
    int i = blockIdx.x * blockDim.x + threadIdx.x;
    if (i < N_NODES) dinv[i] = rsqrtf(dinv[i]);   // deg >= 1 always
}

// ---------------- copy E into output cols [0,128) ----------------
__global__ void copy_embed_kernel(const float* __restrict__ E, float* __restrict__ out) {
    long i = (long)blockIdx.x * blockDim.x + threadIdx.x;  // over N*32 float4 chunks
    if (i >= (long)N_NODES * (EMB / 4)) return;
    long node = i >> 5;
    int  c4   = (int)(i & 31);
    *(float4*)(out + node * OUT_STRIDE + c4 * 4) = ((const float4*)E)[i];
}

// ---------------- WMMA GEMM: Y = X * W^T + b ----------------
// X: [nrows x EMB] f32, row stride xstride (floats). W: [EMB x EMB] f32 (out x in).
// One workgroup = 8 waves; W staged to LDS as f16; each wave: 16 rows x 128 cols.
__global__ __launch_bounds__(256)
void gemm_bias_wmma_kernel(const float* __restrict__ X, long xstride,
                           const float* __restrict__ W, const float* __restrict__ bias,
                           float* __restrict__ Y, int nrows) {
    __shared__ __align__(32) _Float16 Wl[EMB * EMB];   // 32 KB
    int tid = threadIdx.x;
    for (int i = tid; i < EMB * EMB; i += 256) Wl[i] = (_Float16)W[i];
    __syncthreads();

    int wave = tid >> 5;
    int lane = tid & 31;
    int l = lane & 15;            // row within tile (A), col within tile (B/C)
    int h = lane >> 4;            // half-wave selector
    int rowTile = blockIdx.x * 8 + wave;
    if (rowTile * 16 >= nrows) return;   // wave-uniform
    int row0 = rowTile * 16;

    const float* xr = X + (long)(row0 + l) * xstride;

    v8f acc[8];
    #pragma unroll
    for (int t = 0; t < 8; ++t) {
        float bb = bias[t * 16 + l];     // C layout: value(col) depends only on lane&15
        #pragma unroll
        for (int r = 0; r < 8; ++r) acc[t][r] = bb;
    }

    #pragma unroll
    for (int kk = 0; kk < EMB; kk += 32) {
        // A fragment: 16x32 f16. Lane (l,h): halves 0..7 = K[kk+16h .. +7],
        // halves 8..15 = K[kk+16h+8 .. +7]  (contiguous k order, matches B)
        v16h a;
        const float* p1 = xr + kk + 16 * h;
        const float* p2 = xr + kk + 16 * h + 8;
        #pragma unroll
        for (int e = 0; e < 8; ++e) a[e]     = (_Float16)p1[e];
        #pragma unroll
        for (int e = 0; e < 8; ++e) a[8 + e] = (_Float16)p2[e];

        #pragma unroll
        for (int t = 0; t < 8; ++t) {
            // B fragment: 32x16 f16, B[k][n] = W[col0+n][k].
            // Lane (n=l, h): halves e = K[kk+16h+e] -> 16 contiguous f16 of W row.
            const v16h bfrag = *(const v16h*)&Wl[(t * 16 + l) * EMB + kk + 16 * h];
            acc[t] = __builtin_amdgcn_wmma_f32_16x16x32_f16(
                         false, a, false, bfrag, (short)0, acc[t], false, false);
        }
    }

    // C/D layout: VGPR r at lane (l + 16h) = Y[row0 + r + 8h][t*16 + l]
    #pragma unroll
    for (int t = 0; t < 8; ++t) {
        #pragma unroll
        for (int r = 0; r < 8; ++r)
            Y[(long)(row0 + r + 8 * h) * EMB + t * 16 + l] = acc[t][r];
    }
}

// ---------------- aggregation: self-loop init ----------------
__global__ void agg_init_kernel(const float* __restrict__ Y, const float* __restrict__ dinv,
                                float* __restrict__ G) {
    long i = (long)blockIdx.x * blockDim.x + threadIdx.x;  // N*32 float4 chunks
    if (i >= (long)N_NODES * (EMB / 4)) return;
    long node = i >> 5;
    float w = dinv[node]; w = w * w;
    float4 v = ((const float4*)Y)[i];
    ((float4*)G)[i] = make_float4(v.x * w, v.y * w, v.z * w, v.w * w);
}

// ---------------- aggregation: edge scatter (1 wave / edge) ----------------
__global__ __launch_bounds__(256)
void agg_edges_kernel(const float* __restrict__ Y, const float* __restrict__ dinv,
                      const int* __restrict__ src, const int* __restrict__ dst,
                      float* __restrict__ G) {
    int e = blockIdx.x * 8 + (threadIdx.x >> 5);
    if (e >= NUM_EDGES) return;
    int lane = threadIdx.x & 31;
    int s = src[e], d = dst[e];
    float w = dinv[s] * dinv[d];
    float4 v = ((const float4*)(Y + (long)s * EMB))[lane];
    float* gd = G + (long)d * EMB + lane * 4;
    atomicAdd(gd + 0, v.x * w);
    atomicAdd(gd + 1, v.y * w);
    atomicAdd(gd + 2, v.z * w);
    atomicAdd(gd + 3, v.w * w);
}

// ---------------- leaky relu + L2 normalize -> output slice ----------------
__global__ __launch_bounds__(256)
void lrelu_l2_kernel(const float* __restrict__ G, float* __restrict__ out) {
    int node = blockIdx.x * 8 + (threadIdx.x >> 5);
    if (node >= N_NODES) return;
    int lane = threadIdx.x & 31;
    float4 v = ((const float4*)(G + (long)node * EMB))[lane];
    v.x = v.x > 0.0f ? v.x : LRELU_SLOPE * v.x;
    v.y = v.y > 0.0f ? v.y : LRELU_SLOPE * v.y;
    v.z = v.z > 0.0f ? v.z : LRELU_SLOPE * v.z;
    v.w = v.w > 0.0f ? v.w : LRELU_SLOPE * v.w;
    float ss = v.x * v.x + v.y * v.y + v.z * v.z + v.w * v.w;
    #pragma unroll
    for (int o = 16; o >= 1; o >>= 1) ss += __shfl_xor(ss, o, 32);
    float nrm = sqrtf(ss);
    float sc  = 1.0f / fmaxf(nrm, L2_EPS);
    *(float4*)(out + (long)node * OUT_STRIDE + lane * 4) =
        make_float4(v.x * sc, v.y * sc, v.z * sc, v.w * sc);
}

// ---------------- driver ----------------
extern "C" void kernel_launch(void* const* d_in, const int* in_sizes, int n_in,
                              void* d_out, int out_size, void* d_ws, size_t ws_size,
                              hipStream_t stream) {
    const float* E  = (const float*)d_in[0];
    const float* W1 = (const float*)d_in[1];
    const float* b1 = (const float*)d_in[2];
    const float* W2 = (const float*)d_in[3];
    const float* b2 = (const float*)d_in[4];
    const float* W3 = (const float*)d_in[5];
    const float* b3 = (const float*)d_in[6];
    const int*   ei = (const int*)d_in[7];          // [2, E]: row0=src, row1=dst
    const int* srcE = ei;
    const int* dstE = ei + NUM_EDGES;

    float* out = (float*)d_out;

    char* ws = (char*)d_ws;
    size_t oY = (((size_t)N_NODES * 4) + 511) & ~(size_t)511;
    float* dinv = (float*)ws;
    float* Yb   = (float*)(ws + oY);
    float* Gb   = (float*)(ws + oY + (size_t)N_NODES * EMB * 4);

    const int elemBlocks = (N_NODES * (EMB / 4) + 255) / 256;  // 12500
    const int gemmBlocks = ((N_NODES / 16) + 7) / 8;           // 782
    const int edgeBlocks = (NUM_EDGES + 7) / 8;                // 75000
    const int nodeBlocks = (N_NODES + 7) / 8;                  // 12500

    // degrees
    deg_init_kernel <<<(N_NODES + 255) / 256, 256, 0, stream>>>(dinv);
    deg_edges_kernel<<<(NUM_EDGES + 255) / 256, 256, 0, stream>>>(dstE, dinv);
    deg_finish_kernel<<<(N_NODES + 255) / 256, 256, 0, stream>>>(dinv);

    // output cols [0,128) = E
    copy_embed_kernel<<<elemBlocks, 256, 0, stream>>>(E, out);

    const float* Xin[3]    = { E, out + 128, out + 256 };
    const long   Xstride[3]= { EMB, OUT_STRIDE, OUT_STRIDE };
    const float* Ws[3]     = { W1, W2, W3 };
    const float* bs[3]     = { b1, b2, b3 };

    for (int layer = 0; layer < 3; ++layer) {
        gemm_bias_wmma_kernel<<<gemmBlocks, 256, 0, stream>>>(
            Xin[layer], Xstride[layer], Ws[layer], bs[layer], Yb, N_NODES);
        agg_init_kernel <<<elemBlocks, 256, 0, stream>>>(Yb, dinv, Gb);
        agg_edges_kernel<<<edgeBlocks, 256, 0, stream>>>(Yb, dinv, srcE, dstE, Gb);
        lrelu_l2_kernel <<<nodeBlocks, 256, 0, stream>>>(Gb, out + 128 * (layer + 1));
    }
}
